// Encoder_36790689858290
// MI455X (gfx1250) — matrix operand order repaired and checked by reading, not compiled, not run on hardware
//
#include <hip/hip_runtime.h>
#include <hip/hip_bf16.h>

// ---------------------------------------------------------------------------
// Problem constants: N=4, L=1024, C=S=256, d=512, vd=128, LW=600, W=64.
// NL = N*L = 4096, NLW = N*LW = 2400, OUTD = 384.
// ---------------------------------------------------------------------------
#define NL    4096
#define NLW   2400
#define DDIM  512
#define VD    128
#define WWIN  64
#define LWC   600
#define OUTD  384

typedef __attribute__((ext_vector_type(16))) __bf16        v16bf;
typedef __attribute__((ext_vector_type(8)))  float         v8f;
typedef __attribute__((ext_vector_type(4)))  unsigned int  u32x4;
typedef __attribute__((ext_vector_type(4)))  int           v4i;

union Frag {
  v16bf bf;
  u32x4 q[2];
};

__device__ __forceinline__ unsigned short f2bf(float f) {
  unsigned int u = __float_as_uint(f);
  unsigned int r = u + 0x7FFFu + ((u >> 16) & 1u);   // round-to-nearest-even
  return (unsigned short)(r >> 16);
}

// ---------------------------------------------------------------------------
// gfx1250 async global->LDS copy (16B per lane) + ASYNCcnt wait.
// Builtin if the toolchain declares it, else raw CDNA5 ISA via inline asm.
// Probe-learned signature: (v4i addrspace(1)* src, v4i addrspace(3)* dst,
//                           imm int offset, imm int cpol)
// ---------------------------------------------------------------------------
#if defined(__has_builtin)
# if __has_builtin(__builtin_amdgcn_global_load_async_to_lds_b128)
#  define ATHENA_HAS_ASYNC_BUILTIN 1
# endif
# if __has_builtin(__builtin_amdgcn_s_wait_asynccnt)
#  define ATHENA_HAS_WAITASYNC_BUILTIN 1
# endif
#endif

__device__ __forceinline__ void async_cp16(unsigned lds_off, const void* g) {
#ifdef ATHENA_HAS_ASYNC_BUILTIN
  __builtin_amdgcn_global_load_async_to_lds_b128(
      (__attribute__((address_space(1))) v4i*)(size_t)g,
      (__attribute__((address_space(3))) v4i*)(size_t)lds_off, 0, 0);
#else
  // VDST = per-lane LDS byte address, VADDR = per-lane 64-bit global address
  asm volatile("global_load_async_to_lds_b128 %0, %1, off"
               :: "v"(lds_off), "v"(g) : "memory");
#endif
}

__device__ __forceinline__ void wait_async0() {
#ifdef ATHENA_HAS_WAITASYNC_BUILTIN
  __builtin_amdgcn_s_wait_asynccnt(0);
#else
  asm volatile("s_wait_asynccnt 0x0" ::: "memory");
#endif
}

// ---------------------------------------------------------------------------
// LDS fragment loads (documented 16-bit WMMA VGPR layouts).
// A 16x32: lane(0..15)=M, v0..3 K=kh+[0..7], v4..7 K=kh+16+[0..7], kh=(lane>>4)*8
// B 32x16 from BT rows: lane(0..15)=N, v0..3 K=kh+[0..7], v4..7 K=kh+16+[0..7], kh=(lane>>4)*8
// ---------------------------------------------------------------------------
__device__ __forceinline__ v16bf lds_afrag(const unsigned short* row32, int lane) {
  const unsigned short* p = row32 + ((lane >> 4) << 3);
  Frag f;
  f.q[0] = *(const u32x4*)(p);
  f.q[1] = *(const u32x4*)(p + 16);
  return f.bf;
}
__device__ __forceinline__ v16bf lds_bfrag(const unsigned short* row32, int lane) {
  const unsigned short* p = row32 + ((lane >> 4) << 3);
  Frag f;
  f.q[0] = *(const u32x4*)(p);
  f.q[1] = *(const u32x4*)(p + 16);
  return f.bf;
}

// ---------------------------------------------------------------------------
// WMMA bf16 GEMM with double-buffered async LDS staging.
//   C(MxN, f32) = A(MxK) * BT(NxK)^T [+ bias], optional bf16 mirror Cb.
// 256 threads = 8 waves. Block tile 128x128, K-step 32.
// Wave (wm=wv>>2 in {0,1}, wn=wv&3 in {0..3}) computes 64x32 = 4x2 WMMA tiles.
// LDS: As/Bs 2 x 128x32 bf16 = 32 KB total.
// ---------------------------------------------------------------------------
__global__ __launch_bounds__(256) void gemm_bf16_async(
    const unsigned short* __restrict__ A, int lda,
    const unsigned short* __restrict__ BT, int ldb,
    float* __restrict__ C, int ldc,
    unsigned short* __restrict__ Cb,       // may be nullptr
    const float* __restrict__ bias,        // may be nullptr
    int M, int N, int K)
{
  __shared__ unsigned short As[2][128 * 32];
  __shared__ unsigned short Bs[2][128 * 32];

  const int tid  = threadIdx.x;
  const int lane = tid & 31;
  const int wv   = tid >> 5;
  const int wm   = wv >> 2;            // 0..1  -> 64-row slab
  const int wn   = wv & 3;             // 0..3  -> 32-col slab
  const int m0   = blockIdx.y * 128;
  const int n0   = blockIdx.x * 128;

  // staging: 512 16B-chunks per tile; thread t owns chunks t and t+256
  const int c0 = tid, c1 = tid + 256;
  const int r0 = c0 >> 2, o0 = (c0 & 3) << 3;
  const int r1 = c1 >> 2, o1 = (c1 & 3) << 3;
  const unsigned short* gA0 = A  + (size_t)min(m0 + r0, M - 1) * lda + o0;
  const unsigned short* gA1 = A  + (size_t)min(m0 + r1, M - 1) * lda + o1;
  const unsigned short* gB0 = BT + (size_t)min(n0 + r0, N - 1) * ldb + o0;
  const unsigned short* gB1 = BT + (size_t)min(n0 + r1, N - 1) * ldb + o1;
  const unsigned lA0 = (unsigned)(size_t)&As[0][r0 * 32 + o0];
  const unsigned lA1 = (unsigned)(size_t)&As[0][r1 * 32 + o1];
  const unsigned lB0 = (unsigned)(size_t)&Bs[0][r0 * 32 + o0];
  const unsigned lB1 = (unsigned)(size_t)&Bs[0][r1 * 32 + o1];
  const unsigned bufB = 128 * 32 * 2;  // bytes per buffer

  v8f acc[4][2] = {};

  auto stage = [&](int buf, int ks) {
    unsigned bo = (unsigned)buf * bufB;
    async_cp16(lA0 + bo, gA0 + ks);
    async_cp16(lA1 + bo, gA1 + ks);
    async_cp16(lB0 + bo, gB0 + ks);
    async_cp16(lB1 + bo, gB1 + ks);
  };

  stage(0, 0);
  int buf = 0;
  for (int ks = 0; ks < K; ks += 32, buf ^= 1) {
    wait_async0();          // our stage-fill for buf has landed
    __syncthreads();        // everyone's fills visible; prior reads of buf^1 done
    if (ks + 32 < K) stage(buf ^ 1, ks + 32);

    v16bf af[4], bfr[2];
    #pragma unroll
    for (int tm = 0; tm < 4; ++tm)
      af[tm] = lds_afrag(&As[buf][(wm * 64 + tm * 16 + (lane & 15)) * 32], lane);
    #pragma unroll
    for (int tn = 0; tn < 2; ++tn)
      bfr[tn] = lds_bfrag(&Bs[buf][(wn * 32 + tn * 16 + (lane & 15)) * 32], lane);

    #pragma unroll
    for (int tm = 0; tm < 4; ++tm)
      #pragma unroll
      for (int tn = 0; tn < 2; ++tn)
        acc[tm][tn] = __builtin_amdgcn_wmma_f32_16x16x32_bf16(
            false, af[tm], false, bfr[tn], (short)0, acc[tm][tn], false, false);

    __syncthreads();        // all waves done reading buf before it is refilled
  }

  // C/D layout: VGPR r -> M = 8*(lane>>4)+r, N = lane&15 (per 16x16 tile)
  const int mofs = (lane >> 4) << 3;
  const int nofs = lane & 15;
  #pragma unroll
  for (int tm = 0; tm < 4; ++tm)
    #pragma unroll
    for (int tn = 0; tn < 2; ++tn)
      #pragma unroll
      for (int r = 0; r < 8; ++r) {
        int m  = m0 + wm * 64 + tm * 16 + mofs + r;
        int nn = n0 + wn * 32 + tn * 16 + nofs;
        if (m < M && nn < N) {
          float v = acc[tm][tn][r];
          if (bias) v += bias[nn];
          C[(size_t)m * ldc + nn] = v;
          if (Cb) Cb[(size_t)m * ldc + nn] = f2bf(v);
        }
      }
}

// ---------------------------------------------------------------------------
// Prep kernels: bf16 conversions + transposes
// ---------------------------------------------------------------------------
__global__ void cvt_bf16_kernel(const float* __restrict__ src,
                                unsigned short* __restrict__ dst, int n) {
  int i = blockIdx.x * blockDim.x + threadIdx.x;
  if (i < n) dst[i] = f2bf(src[i]);
}

__global__ __launch_bounds__(256) void prep_h_kernel(
    const float* __restrict__ x, const float* __restrict__ sp,
    unsigned short* __restrict__ hb, unsigned short* __restrict__ hT) {
  int row = blockIdx.x;
  int c   = threadIdx.x;             // 0..255
  unsigned short xb = f2bf(x[(size_t)row * 256 + c]);
  unsigned short sb = f2bf(sp[(size_t)row * 256 + c]);
  hb[(size_t)row * DDIM + c]        = xb;
  hb[(size_t)row * DDIM + 256 + c]  = sb;
  hT[(size_t)c * NL + row]          = xb;
  hT[(size_t)(256 + c) * NL + row]  = sb;
}

__global__ void fill_space_kernel(const float* __restrict__ sp,
                                  unsigned short* __restrict__ out2T) {
  int i = blockIdx.x * blockDim.x + threadIdx.x;
  if (i >= NL * 256) return;
  int row = i >> 8, s = i & 255;
  out2T[(size_t)(VD + s) * NL + row] = f2bf(sp[i]);
}

// ---------------------------------------------------------------------------
// Windowed gather attention.  One wave per (n,l); lane owns w=lane, lane+32.
// q row staged in LDS once; wave32 shuffle softmax; coalesced V pass.
// ---------------------------------------------------------------------------
__global__ __launch_bounds__(256) void attn_kernel(
    const float* __restrict__ Q, const float* __restrict__ Kf,
    const float* __restrict__ Vf, const int* __restrict__ mask,
    unsigned short* __restrict__ Ob) {
  __shared__ float4 s_q[8][DDIM / 4];
  __shared__ float  s_attn[8][WWIN];
  __shared__ int    s_row[8][WWIN];
  const int lane = threadIdx.x & 31;
  const int wv   = threadIdx.x >> 5;
  const int idx  = blockIdx.x * 8 + wv;        // 0..4095
  const int n    = idx >> 10;

  const float4* qg = (const float4*)(Q + (size_t)idx * DDIM);
  for (int j = lane; j < DDIM / 4; j += 32) s_q[wv][j] = qg[j];
  __syncthreads();

  float e[2]; int rw[2];
  #pragma unroll
  for (int t = 0; t < 2; ++t) {
    int w  = lane + 32 * t;
    int mi = mask[(size_t)idx * WWIN + w];
    if (mi == LWC) { e[t] = -1e20f; rw[t] = -1; }
    else {
      int r = n * LWC + mi;
      rw[t] = r;
      const float4* kp = (const float4*)(Kf + (size_t)r * DDIM);
      float s = 0.f;
      #pragma unroll 4
      for (int k = 0; k < DDIM / 4; ++k) {
        float4 a = s_q[wv][k], b = kp[k];
        s = fmaf(a.x, b.x, s); s = fmaf(a.y, b.y, s);
        s = fmaf(a.z, b.z, s); s = fmaf(a.w, b.w, s);
      }
      e[t] = s;
    }
  }
  float mx = fmaxf(e[0], e[1]);
  #pragma unroll
  for (int d = 16; d; d >>= 1) mx = fmaxf(mx, __shfl_xor(mx, d, 32));
  float p0 = __expf(e[0] - mx), p1 = __expf(e[1] - mx);
  float sm = p0 + p1;
  #pragma unroll
  for (int d = 16; d; d >>= 1) sm += __shfl_xor(sm, d, 32);
  float inv = 1.0f / sm;
  s_attn[wv][lane]      = p0 * inv;
  s_attn[wv][lane + 32] = p1 * inv;
  s_row[wv][lane]       = rw[0];
  s_row[wv][lane + 32]  = rw[1];
  __syncthreads();

  float o0 = 0.f, o1 = 0.f, o2 = 0.f, o3 = 0.f;
  for (int w = 0; w < WWIN; ++w) {
    int r = s_row[wv][w];              // uniform per wave
    if (r < 0) continue;
    float a = s_attn[wv][w];
    const float* vp = Vf + (size_t)r * VD;
    o0 = fmaf(a, vp[lane],      o0);
    o1 = fmaf(a, vp[lane + 32], o1);
    o2 = fmaf(a, vp[lane + 64], o2);
    o3 = fmaf(a, vp[lane + 96], o3);
  }
  unsigned short* op = Ob + (size_t)idx * VD;
  op[lane]      = f2bf(o0);
  op[lane + 32] = f2bf(o1);
  op[lane + 64] = f2bf(o2);
  op[lane + 96] = f2bf(o3);
}

// ---------------------------------------------------------------------------
// Residual + LayerNorm; writes blk transposed into out2T rows 0..127 (bf16).
// ---------------------------------------------------------------------------
__global__ __launch_bounds__(256) void ln_kernel(
    const float* __restrict__ o2, const float* __restrict__ x,
    const float* __restrict__ g, const float* __restrict__ b,
    unsigned short* __restrict__ out2T) {
  const int lane = threadIdx.x & 31;
  const int wv   = threadIdx.x >> 5;
  const int row  = blockIdx.x * 8 + wv;
  float v[4], s = 0.f;
  #pragma unroll
  for (int i = 0; i < 4; ++i) {
    int c = lane + 32 * i;
    v[i] = o2[(size_t)row * VD + c] + x[(size_t)row * 256 + c];
    s += v[i];
  }
  #pragma unroll
  for (int d = 16; d; d >>= 1) s += __shfl_xor(s, d, 32);
  float mu = s * (1.0f / VD);
  float var = 0.f;
  #pragma unroll
  for (int i = 0; i < 4; ++i) { float dd = v[i] - mu; var += dd * dd; }
  #pragma unroll
  for (int d = 16; d; d >>= 1) var += __shfl_xor(var, d, 32);
  float invs = rsqrtf(var * (1.0f / VD) + 1e-5f);
  #pragma unroll
  for (int i = 0; i < 4; ++i) {
    int c = lane + 32 * i;
    out2T[(size_t)c * NL + row] = f2bf((v[i] - mu) * invs * g[c] + b[c]);
  }
}

// ---------------------------------------------------------------------------
// Host-side orchestration
// ---------------------------------------------------------------------------
static inline size_t alignup(size_t v) { return (v + 255) & ~(size_t)255; }

extern "C" void kernel_launch(void* const* d_in, const int* in_sizes, int n_in,
                              void* d_out, int out_size, void* d_ws, size_t ws_size,
                              hipStream_t stream) {
  (void)in_sizes; (void)n_in; (void)out_size; (void)ws_size;
  const float* x    = (const float*)d_in[0];   // (4,1024,256)
  const int*   mask = (const int*)  d_in[1];   // (4,1024,64)
  const float* ds   = (const float*)d_in[2];   // (2400,4096)
  const float* sp   = (const float*)d_in[4];   // (4,1024,256)
  const float* Wv   = (const float*)d_in[7];   // (128,128)
  const float* Wk   = (const float*)d_in[8];   // (512,512)
  const float* Wq   = (const float*)d_in[9];   // (512,512)
  const float* Wo   = (const float*)d_in[10];  // (128,128)
  const float* bo   = (const float*)d_in[11];  // (128,)
  const float* lng  = (const float*)d_in[12];  // (128,)
  const float* lnb  = (const float*)d_in[13];  // (128,)

  char* ws = (char*)d_ws;
  size_t off = 0;
  auto take = [&](size_t bytes) { char* p = ws + off; off = alignup(off + bytes); return p; };

  unsigned short* hb     = (unsigned short*)take((size_t)NL   * DDIM * 2);
  unsigned short* hT     = (unsigned short*)take((size_t)DDIM * NL   * 2);
  unsigned short* dsb    = (unsigned short*)take((size_t)NLW  * NL   * 2);
  unsigned short* wk_b   = (unsigned short*)take((size_t)DDIM * DDIM * 2);
  unsigned short* wq_b   = (unsigned short*)take((size_t)DDIM * DDIM * 2);
  unsigned short* wv_b   = (unsigned short*)take((size_t)VD * VD * 2);
  unsigned short* wo_b   = (unsigned short*)take((size_t)VD * VD * 2);
  float*          xnew_f = (float*)         take((size_t)NLW * DDIM * 4);
  unsigned short* xnew_b = (unsigned short*)take((size_t)NLW * DDIM * 2);
  float*          kf     = (float*)         take((size_t)NLW * DDIM * 4);
  float*          qf     = (float*)         take((size_t)NL  * DDIM * 4);
  float*          vf     = (float*)         take((size_t)NLW * VD * 4);
  unsigned short* ob     = (unsigned short*)take((size_t)NL * VD * 2);
  float*          o2f    = (float*)         take((size_t)NL * VD * 4);
  unsigned short* out2T  = (unsigned short*)take((size_t)OUTD * NL * 2);

  // --- prep: bf16 conversions + transposes ---------------------------------
  prep_h_kernel<<<NL, 256, 0, stream>>>(x, sp, hb, hT);
  {
    int n = NLW * NL;
    cvt_bf16_kernel<<<(n + 255) / 256, 256, 0, stream>>>(ds, dsb, n);
    cvt_bf16_kernel<<<(DDIM * DDIM + 255) / 256, 256, 0, stream>>>(Wk, wk_b, DDIM * DDIM);
    cvt_bf16_kernel<<<(DDIM * DDIM + 255) / 256, 256, 0, stream>>>(Wq, wq_b, DDIM * DDIM);
    cvt_bf16_kernel<<<(VD * VD + 255) / 256, 256, 0, stream>>>(Wv, wv_b, VD * VD);
    cvt_bf16_kernel<<<(VD * VD + 255) / 256, 256, 0, stream>>>(Wo, wo_b, VD * VD);
  }

  // --- GEMM1: x_new = ds @ h   (2400 x 512, K=4096) ------------------------
  gemm_bf16_async<<<dim3(DDIM / 128, (NLW + 127) / 128), 256, 0, stream>>>(
      dsb, NL, hT, NL, xnew_f, DDIM, xnew_b, nullptr, NLW, DDIM, NL);

  // --- k = x_new @ Wk^T  (2400 x 512, K=512) -------------------------------
  gemm_bf16_async<<<dim3(DDIM / 128, (NLW + 127) / 128), 256, 0, stream>>>(
      xnew_b, DDIM, wk_b, DDIM, kf, DDIM, nullptr, nullptr, NLW, DDIM, DDIM);

  // --- q = h @ Wq^T  (4096 x 512, K=512) -----------------------------------
  gemm_bf16_async<<<dim3(DDIM / 128, NL / 128), 256, 0, stream>>>(
      hb, DDIM, wq_b, DDIM, qf, DDIM, nullptr, nullptr, NL, DDIM, DDIM);

  // --- v = x_new[:, :128] @ Wv^T  (2400 x 128, K=128) ----------------------
  gemm_bf16_async<<<dim3(VD / 128, (NLW + 127) / 128), 256, 0, stream>>>(
      xnew_b, DDIM, wv_b, VD, vf, VD, nullptr, nullptr, NLW, VD, VD);

  // --- windowed gather attention ------------------------------------------
  attn_kernel<<<NL / 8, 256, 0, stream>>>(qf, kf, vf, mask, ob);

  // --- o2 = o @ Wo^T + bo  (4096 x 128, K=128) -----------------------------
  gemm_bf16_async<<<dim3(VD / 128, NL / 128), 256, 0, stream>>>(
      ob, VD, wo_b, VD, o2f, VD, nullptr, bo, NL, VD, VD);

  // --- blk = LN(o2 + x[..., :128]); out2T = [blk; space_pos]^T -------------
  ln_kernel<<<NL / 8, 256, 0, stream>>>(o2f, x, lng, lnb, out2T);
  fill_space_kernel<<<(NL * 256 + 255) / 256, 256, 0, stream>>>(sp, out2T);

  // --- GEMM3: out = ds @ out2  (2400 x 384, K=4096) -> d_out ---------------
  gemm_bf16_async<<<dim3(OUTD / 128, (NLW + 127) / 128), 256, 0, stream>>>(
      dsb, NL, out2T, NL, (float*)d_out, OUTD, nullptr, nullptr, NLW, OUTD, NL);
}
